// Head_67671504716005
// MI455X (gfx1250) — matrix operand order repaired and checked by reading, not compiled, not run on hardware
//
#include <hip/hip_runtime.h>

typedef __attribute__((ext_vector_type(16))) _Float16 v16h;
typedef __attribute__((ext_vector_type(8)))  _Float16 v8h;
typedef __attribute__((ext_vector_type(8)))  float    v8f;
typedef __attribute__((ext_vector_type(4)))  unsigned v4u;
typedef __attribute__((ext_vector_type(8)))  unsigned v8u;

#define WMMA_F32_F16(a, b, c) \
  __builtin_amdgcn_wmma_f32_16x16x32_f16(false, (a), false, (b), (short)0, (c), false, false)

static __device__ __forceinline__ v16h cat8(v8h lo, v8h hi) {
  return __builtin_shufflevector(lo, hi, 0,1,2,3,4,5,6,7,8,9,10,11,12,13,14,15);
}

static __device__ __forceinline__ void wait_tensorcnt0() {
#if __has_builtin(__builtin_amdgcn_s_wait_tensorcnt)
  __builtin_amdgcn_s_wait_tensorcnt(0);
#else
  asm volatile("s_wait_tensorcnt 0x0" ::: "memory");
#endif
}

constexpr int Bb = 4;
constexpr int Tt = 4096;
constexpr int Cc = 1024;
constexpr int Hh = 128;

// ---------------------------------------------------------------------------
// Kernel 1: QKV projection  X[BT,C] @ W[C,H] -> f16 Q/K/V in workspace.
// 128-row M tile per block, 8 waves * 16 rows; K staged in 32-chunks via LDS.
// LDS row stride 40 halfs: 16B-aligned v8h fragment loads, conflict-free.
// ---------------------------------------------------------------------------
constexpr int XSTR = 40;

__global__ __launch_bounds__(256, 1) void qkv_proj(
    const float* __restrict__ X,  const float* __restrict__ Wq,
    const float* __restrict__ Wk, const float* __restrict__ Wv,
    _Float16* __restrict__ Qh, _Float16* __restrict__ Kh, _Float16* __restrict__ Vh)
{
  __shared__ __align__(16) _Float16 Xs[128 * XSTR];
  __shared__ __align__(16) _Float16 Wt[128 * XSTR];
  const int tid  = threadIdx.x;
  const int lane = tid & 31;
  const int wave = tid >> 5;
  const int row0 = blockIdx.x * 128;

  const float* Wmats[3] = {Wq, Wk, Wv};
  _Float16*    Omats[3] = {Qh, Kh, Vh};

#pragma unroll
  for (int mat = 0; mat < 3; ++mat) {
    const float* __restrict__ W = Wmats[mat];
    v8f acc[8] = {};
    for (int kc = 0; kc < Cc; kc += 32) {
      { // stage X[row0..+128][kc..+32] -> Xs (f16)
        const int c = tid & 31, r0 = tid >> 5;
#pragma unroll
        for (int i = 0; i < 16; ++i) {
          const int r = r0 + i * 8;
          Xs[r * XSTR + c] = (_Float16)X[(size_t)(row0 + r) * Cc + kc + c];
        }
      }
      { // stage W[kc..+32][:] transposed -> Wt[h][k]
        const int h = tid & 127, k0 = tid >> 7;
#pragma unroll
        for (int i = 0; i < 16; ++i) {
          const int k = k0 + i * 2;
          Wt[h * XSTR + k] = (_Float16)W[(size_t)(kc + k) * Hh + h];
        }
      }
      __syncthreads();
      const int ar = wave * 16 + (lane & 15);
      const int ak = (lane >> 4) * 8;
      const v16h a = cat8(*(const v8h*)&Xs[ar * XSTR + ak],
                          *(const v8h*)&Xs[ar * XSTR + 16 + ak]);
      const int bc = lane & 15, bk = (lane >> 4) * 8;
#pragma unroll
      for (int nt = 0; nt < 8; ++nt) {
        const _Float16* bp = &Wt[(nt * 16 + bc) * XSTR + bk];
        const v16h b = cat8(*(const v8h*)bp, *(const v8h*)(bp + 16));
        acc[nt] = WMMA_F32_F16(a, b, acc[nt]);
      }
      __syncthreads();
    }
    const float sc = (mat == 0) ? 0.08838834764831845f : 1.0f; // fold 1/sqrt(H) into Q
    _Float16* __restrict__ O = Omats[mat];
    const int cw = lane & 15, rw = (lane >> 4) * 8;
#pragma unroll
    for (int nt = 0; nt < 8; ++nt)
#pragma unroll
      for (int j = 0; j < 8; ++j)
        O[(size_t)(row0 + wave * 16 + rw + j) * Hh + nt * 16 + cw] =
            (_Float16)(acc[nt][j] * sc);
  }
}

// ---------------------------------------------------------------------------
// Kernel 2: causal flash attention, double-buffered KV chunks of 64.
//  - K chunk staged by the Tensor Data Mover (tensor_load_to_lds), with the
//    D# pad fields producing the conflict-free 136-half row stride directly.
//  - V chunk loaded as v8h regs before compute, transposed into LDS after.
//  - P re-laid out (C->A fragment) through wave-private LDS.
// ---------------------------------------------------------------------------
constexpr int KSTR = 136; // 128 halfs/row + 4-DWORD TDM pad
constexpr int VSTR = 72;  // 128 x 64 V^T chunk
constexpr int PSTR = 72;  // 16 x 64 per-wave P tile

__global__ __launch_bounds__(256, 1) void attn(
    const _Float16* __restrict__ Qh, const _Float16* __restrict__ Kh,
    const _Float16* __restrict__ Vh, float* __restrict__ Out)
{
  __shared__ __align__(16) _Float16 Ks[2][64 * KSTR];
  __shared__ __align__(16) _Float16 Vt[2][128 * VSTR];
  __shared__ __align__(16) _Float16 Ps[8 * 16 * PSTR];

  const int tid  = threadIdx.x;
  const int lane = tid & 31;
  const int wave = tid >> 5;
  const int b  = blockIdx.x >> 5;   // T/128 = 32 q-tiles per batch
  const int q0 = (blockIdx.x & 31) * 128;
  const size_t base = (size_t)b * Tt * Hh;

  const int mycol  = lane & 15;
  const int myrow0 = (lane >> 4) << 3;   // 0 or 8 (C-layout row offset)
  const int aoff   = (lane >> 4) * 8;    // A-layout K offset

  // ---- TDM issue: K chunk [64 rows][128 halfs] -> Ks[buf], stride 136 halfs
  auto tdmK = [&](int kv0, int buf) {
    const unsigned long long ga =
        (unsigned long long)(size_t)(const void*)(Kh + base + (size_t)kv0 * Hh);
    const unsigned lds = (unsigned)(size_t)(const void*)&Ks[buf][0];
    v4u g0;
    g0[0] = 1u;                                   // count=1, is_restore=0
    g0[1] = lds;                                  // lds_addr
    g0[2] = (unsigned)ga;                         // global_addr[31:0]
    g0[3] = (unsigned)((ga >> 32) & 0x1FFFFFFu) | (2u << 30); // addr[56:32]|type=2
    v8u g1;
    g1[0] = (1u << 16)      // data_size = 2 bytes
          | (1u << 20)      // pad_enable
          | (5u << 22)      // pad_interval: 64 DWORDs (= 128-half row)
          | (3u << 25);     // pad_amount: 4 DWORDs (= 8 halfs)
    g1[1] = (128u << 16);   // tensor_dim0 = 128
    g1[2] = (64u  << 16);   // tensor_dim1 = 64
    g1[3] = (128u << 16);   // tile_dim0   = 128
    g1[4] = 64u;            // tile_dim1   = 64 (tile_dim2 = 0)
    g1[5] = 128u;           // tensor_dim0_stride = 128 (lo32)
    g1[6] = 0u;             // dim0_stride hi16 | dim1_stride lo16
    g1[7] = 0u;             // dim1_stride hi32
    asm volatile("tensor_load_to_lds %0, %1" :: "s"(g0), "s"(g1) : "memory");
  };

  // ---- V chunk: vectorized global load into regs / transposed LDS store
  v8h vreg[4];
  auto loadV = [&](int kv0) {
#pragma unroll
    for (int i = 0; i < 4; ++i) {
      const int vid = tid + i * 256;
      const int n  = vid >> 4;         // 0..63
      const int hc = (vid & 15) * 8;   // 0..120
      vreg[i] = *(const v8h*)(Vh + base + (size_t)(kv0 + n) * Hh + hc);
    }
  };
  auto storeV = [&](int buf) {
#pragma unroll
    for (int i = 0; i < 4; ++i) {
      const int vid = tid + i * 256;
      const int n  = vid >> 4;
      const int hc = (vid & 15) * 8;
#pragma unroll
      for (int e = 0; e < 8; ++e)
        Vt[buf][(hc + e) * VSTR + n] = vreg[i][e];
    }
  };

  // ---- Q strip as A fragments (16 x 128, 4 k-steps); Q pre-scaled by 1/sqrt(H)
  const int qrow = q0 + wave * 16 + mycol;
  v16h aq[4];
#pragma unroll
  for (int ks = 0; ks < 4; ++ks) {
    const _Float16* qp = Qh + base + (size_t)qrow * Hh + ks * 32 + aoff;
    aq[ks] = cat8(*(const v8h*)qp, *(const v8h*)(qp + 16));
  }

  v8f o[8] = {};
  float m[8], rs[8];
#pragma unroll
  for (int j = 0; j < 8; ++j) { m[j] = -__builtin_inff(); rs[j] = 0.f; }

  const int wqmin = q0 + wave * 16;
  const int wqmax = wqmin + 15;
  const int kvend = q0 + 128;
  constexpr float L2E = 1.44269504088896f;

  // ---- prologue: stage chunk 0 into buffer 0
  loadV(0);
  if (wave == 0) tdmK(0, 0);
  storeV(0);
  if (wave == 0) wait_tensorcnt0();
  __syncthreads();

  int cur = 0;
  for (int kv0 = 0; kv0 < kvend; kv0 += 64, cur ^= 1) {
    const int  kvn      = kv0 + 64;
    const bool havenext = kvn < kvend;
    // async-stage next chunk into the other buffer before computing this one
    if (havenext) {
      loadV(kvn);
      if (wave == 0) tdmK(kvn, cur ^ 1);
    }

    if (kv0 <= wqmax) {  // wave-uniform branch: EXEC all-ones for WMMA
      const _Float16* ksb = Ks[cur];
      const _Float16* vtb = Vt[cur];
      // S = Q K^T  (pre-scaled)
      v8f s[4] = {};
#pragma unroll
      for (int nt = 0; nt < 4; ++nt) {
        const _Float16* kp0 = &ksb[(nt * 16 + mycol) * KSTR + (lane >> 4) * 8];
#pragma unroll
        for (int ks = 0; ks < 4; ++ks) {
          const _Float16* kp = kp0 + ks * 32;
          const v16h bk = cat8(*(const v8h*)kp, *(const v8h*)(kp + 16));
          s[nt] = WMMA_F32_F16(aq[ks], bk, s[nt]);
        }
      }
      // causal mask (diagonal chunks only)
      if (kv0 + 63 > wqmin) {
#pragma unroll
        for (int nt = 0; nt < 4; ++nt) {
          const int coln = kv0 + nt * 16 + mycol;
#pragma unroll
          for (int j = 0; j < 8; ++j) {
            const int rowq = wqmin + myrow0 + j;
            s[nt][j] = (coln > rowq) ? -__builtin_inff() : s[nt][j];
          }
        }
      }
      // online softmax: per C-layout row, reduce across the 16-lane half
#pragma unroll
      for (int j = 0; j < 8; ++j) {
        float mx = fmaxf(fmaxf(s[0][j], s[1][j]), fmaxf(s[2][j], s[3][j]));
        mx = fmaxf(mx, __shfl_xor(mx, 1, 32));
        mx = fmaxf(mx, __shfl_xor(mx, 2, 32));
        mx = fmaxf(mx, __shfl_xor(mx, 4, 32));
        mx = fmaxf(mx, __shfl_xor(mx, 8, 32));
        const float mn   = fmaxf(m[j], mx);
        const float corr = __builtin_amdgcn_exp2f((m[j] - mn) * L2E);
        m[j] = mn;
        float psum = 0.f;
#pragma unroll
        for (int nt = 0; nt < 4; ++nt) {
          const float p = __builtin_amdgcn_exp2f((s[nt][j] - mn) * L2E);
          s[nt][j] = p;
          psum += p;
        }
        psum += __shfl_xor(psum, 1, 32);
        psum += __shfl_xor(psum, 2, 32);
        psum += __shfl_xor(psum, 4, 32);
        psum += __shfl_xor(psum, 8, 32);
        rs[j] = rs[j] * corr + psum;
#pragma unroll
        for (int nt = 0; nt < 8; ++nt) o[nt][j] *= corr;
      }
      // re-layout P (C layout -> A layout) through wave-private LDS
      _Float16* pw = &Ps[wave * 16 * PSTR];
#pragma unroll
      for (int nt = 0; nt < 4; ++nt)
#pragma unroll
        for (int j = 0; j < 8; ++j)
          pw[(myrow0 + j) * PSTR + nt * 16 + mycol] = (_Float16)s[nt][j];
      asm volatile("s_wait_dscnt 0" ::: "memory");
      v16h ap[2];
#pragma unroll
      for (int ks2 = 0; ks2 < 2; ++ks2) {
        const _Float16* pp = &pw[(lane & 15) * PSTR + ks2 * 32 + aoff];
        ap[ks2] = cat8(*(const v8h*)pp, *(const v8h*)(pp + 16));
      }
      // O += P V
#pragma unroll
      for (int nt = 0; nt < 8; ++nt) {
        const _Float16* vp0 = &vtb[(nt * 16 + mycol) * VSTR + (lane >> 4) * 8];
#pragma unroll
        for (int ks2 = 0; ks2 < 2; ++ks2) {
          const _Float16* vp = vp0 + ks2 * 32;
          const v16h bv = cat8(*(const v8h*)vp, *(const v8h*)(vp + 16));
          o[nt] = WMMA_F32_F16(ap[ks2], bv, o[nt]);
        }
      }
    }

    // finish staging of the next chunk, then hand buffers over
    if (havenext) storeV(cur ^ 1);
    if (wave == 0) wait_tensorcnt0();
    __syncthreads();
  }

  // epilogue: Out = O / rowsum
#pragma unroll
  for (int nt = 0; nt < 8; ++nt)
#pragma unroll
    for (int j = 0; j < 8; ++j) {
      const int row = wqmin + myrow0 + j;
      Out[base + (size_t)row * Hh + nt * 16 + mycol] = o[nt][j] / rs[j];
    }
}

// ---------------------------------------------------------------------------
extern "C" void kernel_launch(void* const* d_in, const int* in_sizes, int n_in,
                              void* d_out, int out_size, void* d_ws, size_t ws_size,
                              hipStream_t stream) {
  const float* xd = (const float*)d_in[0];
  const float* Wq = (const float*)d_in[1];
  const float* Wk = (const float*)d_in[2];
  const float* Wv = (const float*)d_in[3];
  float* out = (float*)d_out;

  const size_t nQKV = (size_t)Bb * Tt * Hh;  // 2M elements
  _Float16* Qh = (_Float16*)d_ws;
  _Float16* Kh = Qh + nQKV;
  _Float16* Vh = Kh + nQKV;

  // 1) QKV projection into f16 workspace (Q pre-scaled by 1/sqrt(H))
  qkv_proj<<<dim3((Bb * Tt) / 128), dim3(256), 0, stream>>>(xd, Wq, Wk, Wv, Qh, Kh, Vh);
  // 2) causal flash attention (TDM-staged K, double-buffered)
  attn<<<dim3(Bb * (Tt / 128)), dim3(256), 0, stream>>>(Qh, Kh, Vh, out);
}
